// SpikingNeuralNetwork_72378788872464
// MI455X (gfx1250) — matrix operand order repaired and checked by reading, not compiled, not run on hardware
//
#include <hip/hip_runtime.h>

// ---------------------------------------------------------------------------
// Spiking NN: cur = x@W1+b1 (68.7 GFLOP, dominates), leaky-integrate scan over
// T=256 (sequential per (b,h)), then fc2 on last-step spikes (0.27 GFLOP).
//
// Strategy (MI455X / gfx1250):
//  - Split fp32 x and W1 into bf16 hi+lo; 3x v_wmma_f32_16x16x32_bf16 per
//    K=32 step reproduces fp32 precision (~2^-17 rel err) at bf16 WMMA rate.
//  - Fused GEMM+scan kernel: workgroup (4 wave32) owns a 32b x 32h tile,
//    W1 tile (hi+lo, transposed) staged once in LDS (~129 KB, reused 256x),
//    membrane potential lives in the 8-VGPR WMMA accumulator layout.
//  - fc2 is a tiny exact fp32 scalar kernel.
// ---------------------------------------------------------------------------

typedef __attribute__((ext_vector_type(16))) __bf16 v16bf;
typedef __attribute__((ext_vector_type(8)))  float  v8f;

#define B_SZ 64
#define T_SZ 256
#define K_SZ 1024
#define H_SZ 2048
#define O_SZ 1024

// round-to-nearest-even fp32 -> bf16 (bit trick; NaN irrelevant for this data)
static __device__ __forceinline__ unsigned short f32_to_bf16(float f) {
    unsigned int u = __float_as_uint(f);
    unsigned int lsb = (u >> 16) & 1u;
    u += 0x7fffu + lsb;
    return (unsigned short)(u >> 16);
}
static __device__ __forceinline__ float bf16_to_f32(unsigned short h) {
    return __uint_as_float(((unsigned int)h) << 16);
}

// Split fp32 array into bf16 hi + bf16 lo residual arrays. 4 elems/thread.
__global__ void split_bf16_kernel(const float* __restrict__ src,
                                  unsigned short* __restrict__ hi,
                                  unsigned short* __restrict__ lo, int n4) {
    int i = blockIdx.x * blockDim.x + threadIdx.x;
    if (i >= n4) return;
    float4 v = ((const float4*)src)[i];
    ushort4 H, L;
    H.x = f32_to_bf16(v.x); L.x = f32_to_bf16(v.x - bf16_to_f32(H.x));
    H.y = f32_to_bf16(v.y); L.y = f32_to_bf16(v.y - bf16_to_f32(H.y));
    H.z = f32_to_bf16(v.z); L.z = f32_to_bf16(v.z - bf16_to_f32(H.z));
    H.w = f32_to_bf16(v.w); L.w = f32_to_bf16(v.w - bf16_to_f32(H.w));
    ((ushort4*)hi)[i] = H;
    ((ushort4*)lo)[i] = L;
}

#define LDS_STRIDE 1032  // 1024 + 8 pad, keeps 16B alignment, de-banks reads

// Fused fc1 GEMM + spiking scan. Grid: (H/32, B/32), block: 128 (4 waves).
// Wave w: wr=w>>1 selects 16-row b half, wc=w&1 selects 16-col h half.
__global__ __launch_bounds__(128)
void snn_fused_kernel(const unsigned short* __restrict__ xh,
                      const unsigned short* __restrict__ xl,
                      const unsigned short* __restrict__ w1h,
                      const unsigned short* __restrict__ w1l,
                      const float* __restrict__ b1,
                      float* __restrict__ spk) {
    __shared__ unsigned short Bh[32][LDS_STRIDE];
    __shared__ unsigned short Bl[32][LDS_STRIDE];

    const int tid  = threadIdx.x;
    const int lane = tid & 31;
    const int wave = tid >> 5;
    const int wr = wave >> 1;
    const int wc = wave & 1;
    const int h0 = blockIdx.x * 32;
    const int b0 = blockIdx.y * 32;

    // Stage W1 tile (hi+lo), transposed to column-major-in-K, once per block.
    for (int idx = tid; idx < 32 * K_SZ; idx += 128) {
        int k = idx >> 5;
        int n = idx & 31;
        Bh[n][k] = w1h[(size_t)k * H_SZ + h0 + n];
        Bl[n][k] = w1l[(size_t)k * H_SZ + h0 + n];
    }
    __syncthreads();

    const int mrow    = lane & 15;
    const int hi_half = (lane >> 4) & 1;  // lanes 16..31

    const float bias = b1[h0 + wc * 16 + mrow];

    // A fragment (16-bit A 16x32): lanes 0-15 take K {0..15},
    // lanes 16-31 take K {16..31} of row M = lane&15.
    const size_t arow = (size_t)(b0 + wr * 16 + mrow) * T_SZ * K_SZ;
    const int ka = hi_half ? 16 : 0;

    // B fragment (16-bit B 32x16): lane holds column n=lane&15,
    // K {0..15} (lanes 0-15) or K {16..31} (lanes 16-31), contiguous in LDS.
    const unsigned short* browh = &Bh[wc * 16 + mrow][hi_half ? 16 : 0];
    const unsigned short* browl = &Bl[wc * 16 + mrow][hi_half ? 16 : 0];

    float mem[8];
    float spkv[8];
#pragma unroll
    for (int j = 0; j < 8; ++j) { mem[j] = 0.0f; spkv[j] = 0.0f; }

    const float decay = 0.9f;
    const float omd   = (float)(1.0 - 0.9);  // matches reference's (1.0-DECAY)

    for (int t = 0; t < T_SZ; ++t) {
        v8f c;
#pragma unroll
        for (int j = 0; j < 8; ++j) c[j] = bias;

        const unsigned short* ah_t = xh + arow + (size_t)t * K_SZ + ka;
        const unsigned short* al_t = xl + arow + (size_t)t * K_SZ + ka;

#pragma unroll 2
        for (int k0 = 0; k0 < K_SZ; k0 += 32) {
            union Frag { uint4 u[2]; v16bf v; } Ah, Al, Fbh, Fbl;
            Ah.u[0] = *(const uint4*)(ah_t + k0);
            Ah.u[1] = *(const uint4*)(ah_t + k0 + 8);
            Al.u[0] = *(const uint4*)(al_t + k0);
            Al.u[1] = *(const uint4*)(al_t + k0 + 8);
            Fbh.u[0] = *(const uint4*)(browh + k0);
            Fbh.u[1] = *(const uint4*)(browh + k0 + 8);
            Fbl.u[0] = *(const uint4*)(browl + k0);
            Fbl.u[1] = *(const uint4*)(browl + k0 + 8);
            c = __builtin_amdgcn_wmma_f32_16x16x32_bf16(false, Ah.v, false, Fbh.v,
                                                        (short)0, c, false, false);
            c = __builtin_amdgcn_wmma_f32_16x16x32_bf16(false, Ah.v, false, Fbl.v,
                                                        (short)0, c, false, false);
            c = __builtin_amdgcn_wmma_f32_16x16x32_bf16(false, Al.v, false, Fbh.v,
                                                        (short)0, c, false, false);
        }

        const bool last = (t == T_SZ - 1);
#pragma unroll
        for (int j = 0; j < 8; ++j) {
            float m = mem[j] * decay + c[j] * omd;
            bool fired = m > 1.0f;
            mem[j] = fired ? 0.0f : m;
            if (last) spkv[j] = fired ? 1.0f : 0.0f;
        }
    }

    // C/D layout: VGPR j -> M = j (+8 for lanes 16-31), N = lane&15
#pragma unroll
    for (int j = 0; j < 8; ++j) {
        int row = b0 + wr * 16 + hi_half * 8 + j;
        int col = h0 + wc * 16 + mrow;
        spk[(size_t)row * H_SZ + col] = spkv[j];
    }
}

// fc2: out[b,o] = b2[o] + sum_h spk[b,h] * W2[h,o]   (exact fp32, tiny)
__global__ __launch_bounds__(256)
void fc2_kernel(const float* __restrict__ spk, const float* __restrict__ W2,
                const float* __restrict__ b2, float* __restrict__ out) {
    int o = blockIdx.x * 256 + threadIdx.x;
    int b = blockIdx.y;
    float acc = b2[o];
    const float* s = spk + (size_t)b * H_SZ;
    for (int h = 0; h < H_SZ; ++h)
        acc = fmaf(s[h], W2[(size_t)h * O_SZ + o], acc);
    out[(size_t)b * O_SZ + o] = acc;
}

extern "C" void kernel_launch(void* const* d_in, const int* in_sizes, int n_in,
                              void* d_out, int out_size, void* d_ws, size_t ws_size,
                              hipStream_t stream) {
    const float* x  = (const float*)d_in[0];  // [64,256,1024]
    const float* W1 = (const float*)d_in[1];  // [1024,2048]
    const float* b1 = (const float*)d_in[2];  // [2048]
    const float* W2 = (const float*)d_in[3];  // [2048,1024]
    const float* b2 = (const float*)d_in[4];  // [1024]
    float* out = (float*)d_out;               // [64,1024]

    // Workspace layout (bytes): xh 32M | xl 32M | w1h 4M | w1l 4M | spk 0.5M
    char* ws = (char*)d_ws;
    unsigned short* xh  = (unsigned short*)(ws);
    unsigned short* xl  = (unsigned short*)(ws + (size_t)33554432);
    unsigned short* w1h = (unsigned short*)(ws + (size_t)67108864);
    unsigned short* w1l = (unsigned short*)(ws + (size_t)71303168);
    float*          spk = (float*)(ws + (size_t)75497472);

    const int nx4 = (B_SZ * T_SZ * K_SZ) / 4;  // 4,194,304
    split_bf16_kernel<<<nx4 / 256, 256, 0, stream>>>(x, xh, xl, nx4);
    const int nw4 = (K_SZ * H_SZ) / 4;         // 524,288
    split_bf16_kernel<<<nw4 / 256, 256, 0, stream>>>(W1, w1h, w1l, nw4);

    snn_fused_kernel<<<dim3(H_SZ / 32, B_SZ / 32), 128, 0, stream>>>(
        xh, xl, w1h, w1l, b1, spk);

    fc2_kernel<<<dim3(O_SZ / 256, B_SZ), 256, 0, stream>>>(spk, W2, b2, out);
}
